// MultiHeadAttention_10917806866666
// MI455X (gfx1250) — matrix operand order repaired and checked by reading, not compiled, not run on hardware
//
#include <hip/hip_runtime.h>

// ---------------- problem constants ----------------
#define HIDDEN 1024
#define SEQ    2048
#define BATCH  4
#define HEADS  16
#define HD     64
#define MROWS  (BATCH * SEQ)          // 8192

typedef __attribute__((ext_vector_type(8)))  float   v8f;
typedef __attribute__((ext_vector_type(8)))  __bf16  v8bf;
typedef __attribute__((ext_vector_type(16))) __bf16  v16bf;
typedef unsigned short ushort_t;

// ---------------- helpers ----------------
static __device__ inline ushort_t f2bf(float f) {
  union { float f; unsigned u; } c; c.f = f;
  unsigned u = c.u;
  u += 0x7fffu + ((u >> 16) & 1u);          // round-to-nearest-even
  return (ushort_t)(u >> 16);
}

// two contiguous 16B chunks -> one 16-element bf16 fragment
static __device__ inline v16bf ld_frag(const ushort_t* p0, const ushort_t* p1) {
  union { v16bf v; v8bf h[2]; } u;
  u.h[0] = *reinterpret_cast<const v8bf*>(p0);
  u.h[1] = *reinterpret_cast<const v8bf*>(p1);
  return u.v;
}

static __device__ inline v8f wmma_bf16(v16bf a, v16bf b, v8f c) {
  return __builtin_amdgcn_wmma_f32_16x16x32_bf16(false, a, false, b, (short)0, c,
                                                 false, false);
}

// ---------------- kernel: zero the context-sum accumulator ----------------
__global__ void zero_csum(float* csum) {
  int gid = blockIdx.x * blockDim.x + threadIdx.x;   // 4096 threads
  csum[gid] = 0.0f;
}

// ---------------- kernel: fp32 -> bf16 convert (vector-of-4) ----------------
__global__ void cvt_bf16(const float* __restrict__ src, ushort_t* __restrict__ dst) {
  int gid = blockIdx.x * blockDim.x + threadIdx.x;
  const float4 f4 = reinterpret_cast<const float4*>(src)[gid];
  ushort_t o[4] = { f2bf(f4.x), f2bf(f4.y), f2bf(f4.z), f2bf(f4.w) };
  uint2 pk;
  pk.x = (unsigned)o[0] | ((unsigned)o[1] << 16);
  pk.y = (unsigned)o[2] | ((unsigned)o[3] << 16);
  reinterpret_cast<uint2*>(dst)[gid] = pk;
}

// ---------------- kernel: QKV projection, y = x @ W.T + b ----------------
// grid (256, 2, 3), block 256.  Each wave: 32x64 output tile (8 WMMAs/k-step,
// every B fragment reused twice).
// z = 0 -> Q*0.125 [b,h,t,64] ; z = 1 -> K [b,h,t,64] ; z = 2 -> V^T [b,h,64,t]
__global__ __launch_bounds__(256) void qkv_gemm(
    const ushort_t* __restrict__ xb,
    const ushort_t* __restrict__ wq, const ushort_t* __restrict__ wk,
    const ushort_t* __restrict__ wv,
    const float* __restrict__ bq, const float* __restrict__ bk,
    const float* __restrict__ bv,
    ushort_t* __restrict__ qbuf, ushort_t* __restrict__ kbuf,
    ushort_t* __restrict__ vtbuf)
{
  const int mode = blockIdx.z;
  const ushort_t* W   = (mode == 0) ? wq : (mode == 1) ? wk : wv;
  const float*    bia = (mode == 0) ? bq : (mode == 1) ? bk : bv;
  const float     osc = (mode == 0) ? 0.125f : 1.0f;   // fold softmax scale into Q

  const int lane  = threadIdx.x & 31;
  const int wave  = threadIdx.x >> 5;
  const int l15   = lane & 15;
  const int half  = lane >> 4;
  const int kg    = half * 8;

  const int mbase = blockIdx.x * 32;
  const int nbase = (blockIdx.y * 8 + wave) * 64;

  v8f acc[2][4];
  v8f zz = {};
#pragma unroll
  for (int p = 0; p < 2; ++p)
#pragma unroll
    for (int j = 0; j < 4; ++j) acc[p][j] = zz;

  for (int k0 = 0; k0 < HIDDEN; k0 += 32) {
    const ushort_t* ap0 = xb + (mbase + l15) * HIDDEN + k0 + kg;
    const ushort_t* ap1 = ap0 + 16 * HIDDEN;
    const v16bf a0 = ld_frag(ap0, ap0 + 16);
    const v16bf a1 = ld_frag(ap1, ap1 + 16);
    const int kb2 = k0 + half * 8;
#pragma unroll
    for (int j = 0; j < 4; ++j) {
      const ushort_t* bp = W + (nbase + j * 16 + l15) * HIDDEN + kb2;
      const v16bf bf = ld_frag(bp, bp + 16);
      acc[0][j] = wmma_bf16(a0, bf, acc[0][j]);
      acc[1][j] = wmma_bf16(a1, bf, acc[1][j]);
    }
  }

  const int b  = mbase >> 11;        // mbase / 2048
  const int tb = mbase & 2047;

  if (mode < 2) {
    ushort_t* dst = (mode == 0) ? qbuf : kbuf;
#pragma unroll
    for (int p = 0; p < 2; ++p)
#pragma unroll
      for (int j = 0; j < 4; ++j) {
        const int n  = nbase + j * 16 + l15;
        const float bv0 = bia[n];
        const int h  = n >> 6, hd = n & 63;
        const long rowoff = (long)(b * HEADS + h) * SEQ;
#pragma unroll
        for (int e = 0; e < 8; ++e) {
          const int t = tb + p * 16 + e + 8 * half;
          dst[(rowoff + t) * HD + hd] = f2bf((acc[p][j][e] + bv0) * osc);
        }
      }
  } else {
#pragma unroll
    for (int p = 0; p < 2; ++p)
#pragma unroll
      for (int j = 0; j < 4; ++j) {
        const int n  = nbase + j * 16 + l15;
        const float bv0 = bia[n];
        const int h  = n >> 6, hd = n & 63;
        union { ushort_t s[8]; uint4 u; } pk;
#pragma unroll
        for (int e = 0; e < 8; ++e) pk.s[e] = f2bf(acc[p][j][e] + bv0);
        ushort_t* dst = vtbuf + ((long)(b * HEADS + h) * HD + hd) * SEQ
                              + tb + p * 16 + 8 * half;
        *reinterpret_cast<uint4*>(dst) = pk.u;     // 16B, aligned
      }
  }
}

// ---------------- kernel: flash attention, 16 queries per wave ----------------
// 64 keys per outer step: 8 score WMMAs + 8 PV WMMAs, one shuffle tree per step.
// grid 1024, block 256 (8 waves). csum[b*1024 + h*64 + d] += sum_t O[t,d]
__global__ __launch_bounds__(256) void flash_attn(
    const ushort_t* __restrict__ qbuf, const ushort_t* __restrict__ kbuf,
    const ushort_t* __restrict__ vtbuf, float* __restrict__ csum)
{
  __shared__ __align__(16) ushort_t pls[8][1024];  // 16x64 bf16 P-tile per wave

  const int lane = threadIdx.x & 31;
  const int wave = threadIdx.x >> 5;
  const int l15  = lane & 15;
  const int half = lane >> 4;
  const int kg   = half * 16;

  const int gw = blockIdx.x * 8 + wave;   // 0 .. 8191
  const int bh = gw >> 7;                 // (b*16 + h)
  const int qt = gw & 127;

  // Q fragments (pre-scaled by 1/8; reused for every key step)
  const ushort_t* Qrow = qbuf + ((long)bh * SEQ + qt * 16 + l15) * HD;
  const v16bf qa0 = ld_frag(Qrow + kg,      Qrow + kg + 8);         // d 0..31
  const v16bf qa1 = ld_frag(Qrow + 32 + kg, Qrow + 32 + kg + 8);    // d 32..63

  v8f m, l, acc[4];
#pragma unroll
  for (int e = 0; e < 8; ++e) { m[e] = -3.0e38f; l[e] = 0.0f; }
  v8f zz = {};
  acc[0] = zz; acc[1] = zz; acc[2] = zz; acc[3] = zz;

  ushort_t* P = &pls[wave][0];
  const ushort_t* Kbh = kbuf  + (long)bh * SEQ * HD;
  const ushort_t* Vbh = vtbuf + (long)bh * HD * SEQ;

  for (int kb0 = 0; kb0 < SEQ; kb0 += 64) {
    // ---- S = Q K^T for 16x64 scores (8 WMMAs)
    v8f s[4];
#pragma unroll
    for (int i = 0; i < 4; ++i) {
      const ushort_t* kp = Kbh + (kb0 + 16 * i + l15) * HD + half * 16;
      v8f si = zz;
      si = wmma_bf16(qa0, ld_frag(kp,      kp + 8),  si);
      si = wmma_bf16(qa1, ld_frag(kp + 32, kp + 40), si);
      s[i] = si;
    }

    // ---- online softmax (rows live per-element, cols across 16-lane halves)
    v8f sm;
#pragma unroll
    for (int e = 0; e < 8; ++e)
      sm[e] = fmaxf(fmaxf(s[0][e], s[1][e]), fmaxf(s[2][e], s[3][e]));
#pragma unroll
    for (int off = 1; off <= 8; off <<= 1)
#pragma unroll
      for (int e = 0; e < 8; ++e)
        sm[e] = fmaxf(sm[e], __shfl_xor(sm[e], off, 32));

    v8f al, rs;
    v8f pv[4];
#pragma unroll
    for (int e = 0; e < 8; ++e) {
      const float mn = fmaxf(m[e], sm[e]);
      al[e] = __expf(m[e] - mn);
      m[e]  = mn;
      float r = 0.0f;
#pragma unroll
      for (int i = 0; i < 4; ++i) {
        pv[i][e] = __expf(s[i][e] - mn);
        r += pv[i][e];
      }
      rs[e] = r;
    }
#pragma unroll
    for (int off = 1; off <= 8; off <<= 1)
#pragma unroll
      for (int e = 0; e < 8; ++e)
        rs[e] += __shfl_xor(rs[e], off, 32);
#pragma unroll
    for (int e = 0; e < 8; ++e) l[e] = l[e] * al[e] + rs[e];
#pragma unroll
    for (int j = 0; j < 4; ++j)
#pragma unroll
      for (int e = 0; e < 8; ++e) acc[j][e] *= al[e];

    // ---- transpose P (D-layout -> A-layout) through LDS
#pragma unroll
    for (int i = 0; i < 4; ++i)
#pragma unroll
      for (int e = 0; e < 8; ++e) {
        const int r = e + 8 * half;
        P[r * 64 + i * 16 + l15] = f2bf(pv[i][e]);
      }
    __builtin_amdgcn_wave_barrier();          // keep ds_store before ds_load
    const ushort_t* pr = P + l15 * 64 + kg;
    const v16bf pa0 = ld_frag(pr,      pr + 8);    // keys  0..31 (A layout)
    const v16bf pa1 = ld_frag(pr + 32, pr + 40);   // keys 32..63
    __builtin_amdgcn_wave_barrier();

    // ---- O += P V  (8 WMMAs over Hd = 64, K = 64 keys)
#pragma unroll
    for (int j = 0; j < 4; ++j) {
      const ushort_t* vp = Vbh + (j * 16 + l15) * SEQ + kb0 + half * 16;
      acc[j] = wmma_bf16(pa0, ld_frag(vp,      vp + 8),  acc[j]);
      acc[j] = wmma_bf16(pa1, ld_frag(vp + 32, vp + 40), acc[j]);
    }
  }

  // ---- epilogue: O /= l, sum over the 16 query rows, atomically fold into csum
  v8f il;
#pragma unroll
  for (int e = 0; e < 8; ++e) il[e] = 1.0f / l[e];

  const int b = bh >> 4, h = bh & 15;
#pragma unroll
  for (int j = 0; j < 4; ++j) {
    float part = 0.0f;
#pragma unroll
    for (int e = 0; e < 8; ++e) part += acc[j][e] * il[e];
    part += __shfl_xor(part, 16, 32);         // combine both row halves
    if (half == 0)
      atomicAdd(&csum[(b << 10) + (h << 6) + j * 16 + l15], part);
  }
}

// ---------------- kernel: out = (csum/T) @ Wo.T + bo ----------------
// One wave per output element; lanes strided over k (coalesced float4 loads).
__global__ __launch_bounds__(256) void final_proj(
    const float* __restrict__ csum, const float* __restrict__ Wo,
    const float* __restrict__ bo, float* __restrict__ out)
{
  const int lane = threadIdx.x & 31;
  const int wave = threadIdx.x >> 5;
  const int gw = blockIdx.x * 8 + wave;     // 0 .. 4095
  const int b = gw >> 10, n = gw & 1023;
  const float* wr = Wo + (long)n * HIDDEN;
  const float* cr = csum + b * HIDDEN;
  float acc = 0.0f;
#pragma unroll
  for (int kk = 0; kk < 8; ++kk) {
    const int k = kk * 128 + lane * 4;
    const float4 w4 = *reinterpret_cast<const float4*>(wr + k);
    const float4 c4 = *reinterpret_cast<const float4*>(cr + k);
    acc += w4.x * c4.x + w4.y * c4.y + w4.z * c4.z + w4.w * c4.w;
  }
#pragma unroll
  for (int off = 16; off >= 1; off >>= 1) acc += __shfl_xor(acc, off, 32);
  if (lane == 0) out[gw] = acc * (1.0f / (float)SEQ) + bo[n];
}

// ---------------- host ----------------
extern "C" void kernel_launch(void* const* d_in, const int* in_sizes, int n_in,
                              void* d_out, int out_size, void* d_ws, size_t ws_size,
                              hipStream_t stream) {
  (void)in_sizes; (void)n_in; (void)out_size; (void)ws_size;
  const float* x  = (const float*)d_in[0];
  const float* Wq = (const float*)d_in[1];
  const float* bq = (const float*)d_in[2];
  const float* Wk = (const float*)d_in[3];
  const float* bk = (const float*)d_in[4];
  const float* Wv = (const float*)d_in[5];
  const float* bv = (const float*)d_in[6];
  const float* Wo = (const float*)d_in[7];
  const float* bo = (const float*)d_in[8];
  float* out = (float*)d_out;

  // workspace carve-up (bytes): all 16B aligned; total ~73.4 MB
  char* p = (char*)d_ws;
  ushort_t* xb  = (ushort_t*)p;  p += (size_t)MROWS * HIDDEN * 2;   // 16.78 MB
  ushort_t* wqb = (ushort_t*)p;  p += (size_t)HIDDEN * HIDDEN * 2;  // 2 MB
  ushort_t* wkb = (ushort_t*)p;  p += (size_t)HIDDEN * HIDDEN * 2;
  ushort_t* wvb = (ushort_t*)p;  p += (size_t)HIDDEN * HIDDEN * 2;
  ushort_t* qb  = (ushort_t*)p;  p += (size_t)MROWS * HIDDEN * 2;
  ushort_t* kb  = (ushort_t*)p;  p += (size_t)MROWS * HIDDEN * 2;
  ushort_t* vt  = (ushort_t*)p;  p += (size_t)MROWS * HIDDEN * 2;
  float*    cs  = (float*)p;                                        // 16 KB

  zero_csum<<<16, 256, 0, stream>>>(cs);

  cvt_bf16<<<(MROWS * HIDDEN / 4) / 256, 256, 0, stream>>>(x,  xb);
  cvt_bf16<<<(HIDDEN * HIDDEN / 4) / 256, 256, 0, stream>>>(Wq, wqb);
  cvt_bf16<<<(HIDDEN * HIDDEN / 4) / 256, 256, 0, stream>>>(Wk, wkb);
  cvt_bf16<<<(HIDDEN * HIDDEN / 4) / 256, 256, 0, stream>>>(Wv, wvb);

  qkv_gemm<<<dim3(MROWS / 32, 2, 3), 256, 0, stream>>>(
      xb, wqb, wkb, wvb, bq, bk, bv, qb, kb, vt);

  flash_attn<<<1024, 256, 0, stream>>>(qb, kb, vt, cs);

  final_proj<<<512, 256, 0, stream>>>(cs, Wo, bo, out);
}
